// Reformer_18966575579405
// MI455X (gfx1250) — matrix operand order repaired and checked by reading, not compiled, not run on hardware
//
#include <hip/hip_runtime.h>
#include <hip/hip_bf16.h>
#include <math.h>

// ---------------------------------------------------------------------------
// Types for CDNA5 WMMA (wave32, 16x16x32 bf16 -> f32)
// ---------------------------------------------------------------------------
typedef __attribute__((ext_vector_type(16))) __bf16 bf16x16;
typedef __attribute__((ext_vector_type(8)))  float  f32x8;

__device__ __forceinline__ unsigned short f2bf(float f) {
  unsigned u = __float_as_uint(f);
  unsigned r = 0x7FFFu + ((u >> 16) & 1u);   // round-to-nearest-even
  u += r;
  return (unsigned short)(u >> 16);
}

__device__ __forceinline__ float allsum32(float x) {
  #pragma unroll
  for (int m = 16; m > 0; m >>= 1) x += __shfl_xor(x, m, 32);
  return x;
}

// ---------------------------------------------------------------------------
// Embedding: circular conv1d(k=3, no bias) + sinusoidal PE  (x_mark is zero)
// x: [T=2048, N=16, 16]  -> h: [N,T,256]
// ---------------------------------------------------------------------------
__global__ __launch_bounds__(256) void embed_kernel(
    const float* __restrict__ x, const float* __restrict__ conv_w,
    float* __restrict__ h)
{
  int idx = blockIdx.x * 256 + threadIdx.x;      // 32768*256 threads
  int dm  = idx & 255;
  int row = idx >> 8;                            // row = n*2048 + t
  int t   = row & 2047;
  int n   = row >> 11;
  float acc = 0.f;
  #pragma unroll
  for (int k = 0; k < 3; ++k) {
    int tt = (t + k - 1 + 2048) & 2047;          // wrap padding
    const float* xp = x + ((size_t)tt * 16 + n) * 16;
    const float* w  = conv_w + (size_t)k * 16 * 256 + dm;
    #pragma unroll
    for (int i = 0; i < 16; ++i) acc += xp[i] * w[i * 256];
  }
  int   i2   = dm & ~1;
  float freq = expf(-(float)i2 * (logf(10000.f) / 256.f));
  float arg  = (float)t * freq;
  acc += (dm & 1) ? cosf(arg) : sinf(arg);
  h[(size_t)row * 256 + dm] = acc;
}

// ---------------------------------------------------------------------------
// bf16 WMMA GEMM:  C[M,N] = A[M,K] * B[K,N]  (+bias, +gelu, +residual)
// block = 256 thr (8 waves), tile 128x64; each wave owns a 32x32 sub-tile
// (2x2 WMMA 16x16x32 tiles, A/B fragment reuse). B staged transposed in LDS
// so every fragment is two contiguous ds_load_b128.
// ---------------------------------------------------------------------------
#define LDA 40   // padded row stride (ushorts): 80B, 16B-aligned, bank-spread
#define LDB 40

template <bool BIAS, bool GELU, bool RES>
__global__ __launch_bounds__(256) void gemm_wmma(
    const float* __restrict__ A, const float* __restrict__ B,
    const float* __restrict__ bias, const float* __restrict__ res,
    float* __restrict__ C, int M, int N, int K)
{
  __shared__ unsigned short As[128 * LDA];   // A panel 128x32 (bf16)
  __shared__ unsigned short Bt[64 * LDB];    // B panel transposed: [n][k] 64x32
  const int tid  = threadIdx.x;
  const int wave = tid >> 5, lane = tid & 31;
  const int m0 = blockIdx.x * 128, n0 = blockIdx.y * 64;
  const int mloc = lane & 15, half = lane >> 4;
  const int wm = wave >> 1, wn = wave & 1;   // wave -> (m 32-row, n 32-col)

  f32x8 acc[2][2] = {{{}, {}}, {{}, {}}};

  for (int k0 = 0; k0 < K; k0 += 32) {
    __syncthreads();
    #pragma unroll
    for (int it = 0; it < 4; ++it) {           // A panel 128x32 fp32 -> bf16
      int idx = tid + it * 256;                // 0..1023
      int row = idx >> 3, c4 = idx & 7;
      const float* ap = A + (size_t)(m0 + row) * K + k0 + c4 * 4;
      const float4 f = *reinterpret_cast<const float4*>(ap);
      if (k0 + 32 < K) __builtin_prefetch(ap + 32, 0, 3);
      int b = row * LDA + c4 * 4;
      As[b] = f2bf(f.x); As[b+1] = f2bf(f.y); As[b+2] = f2bf(f.z); As[b+3] = f2bf(f.w);
    }
    #pragma unroll
    for (int it = 0; it < 2; ++it) {           // B panel 32x64 -> transposed LDS
      int idx = tid + it * 256;                // 0..511
      int krow = idx >> 4, c4 = idx & 15;
      const float* bp = B + (size_t)(k0 + krow) * N + n0 + c4 * 4;
      const float4 f = *reinterpret_cast<const float4*>(bp);
      if (k0 + 32 < K) __builtin_prefetch(bp + 32 * N, 0, 3);
      int nb = c4 * 4;
      Bt[(nb + 0) * LDB + krow] = f2bf(f.x);
      Bt[(nb + 1) * LDB + krow] = f2bf(f.y);
      Bt[(nb + 2) * LDB + krow] = f2bf(f.z);
      Bt[(nb + 3) * LDB + krow] = f2bf(f.w);
    }
    __syncthreads();

    union { bf16x16 v; unsigned short u[16]; } af[2], bf[2];
    #pragma unroll
    for (int sm = 0; sm < 2; ++sm) {           // A fragments (16x32 each)
      const int arow = wm * 32 + sm * 16 + mloc;
      #pragma unroll
      for (int e = 0; e < 16; ++e) af[sm].u[e] = As[arow * LDA + 16 * half + e];
    }
    #pragma unroll
    for (int sn = 0; sn < 2; ++sn) {           // B fragments (32x16 each)
      const int ncol = wn * 32 + sn * 16 + mloc;
      #pragma unroll
      for (int e = 0; e < 16; ++e) bf[sn].u[e] = Bt[ncol * LDB + 16 * half + e];
    }
    #pragma unroll
    for (int sm = 0; sm < 2; ++sm)
      #pragma unroll
      for (int sn = 0; sn < 2; ++sn)
        acc[sm][sn] = __builtin_amdgcn_wmma_f32_16x16x32_bf16(
            false, af[sm].v, false, bf[sn].v, (short)0, acc[sm][sn], false, false);
  }

  const int nl = lane & 15;
  #pragma unroll
  for (int sm = 0; sm < 2; ++sm) {
    #pragma unroll
    for (int sn = 0; sn < 2; ++sn) {
      #pragma unroll
      for (int r = 0; r < 8; ++r) {
        int m = m0 + wm * 32 + sm * 16 + r + 8 * half;
        int n = n0 + wn * 32 + sn * 16 + nl;
        float c = acc[sm][sn][r];
        if (BIAS) c += bias[n];
        if (GELU) c = 0.5f * c * (1.0f + erff(c * 0.70710678118654752f));
        if (RES)  c += res[(size_t)m * N + n];
        C[(size_t)m * N + n] = c;
      }
    }
  }
}

// ---------------------------------------------------------------------------
// LSH hashing: bucket = argmax over [R, -R], R = qk_head . rot[:,hash,:]
// qk: [16,2048,256]; rot: [64,4,256]; buckets: [64][4][2048], in [0,512)
// ---------------------------------------------------------------------------
__global__ __launch_bounds__(256) void hash_kernel(
    const float* __restrict__ qk, const float* __restrict__ rot,
    int* __restrict__ buckets)
{
  int idx = blockIdx.x * 256 + threadIdx.x;      // 64*2048
  int bh = idx >> 11, t = idx & 2047;
  int n = bh >> 2, head = bh & 3;
  const float* q = qk + ((size_t)n * 2048 + t) * 256 + head * 64;
  float qr[64];
  #pragma unroll
  for (int d = 0; d < 64; ++d) qr[d] = q[d];
  for (int hsh = 0; hsh < 4; ++hsh) {
    float best = -3.4e38f; int bi = 0;
    for (int i = 0; i < 256; ++i) {
      float s = 0.f;
      #pragma unroll
      for (int d = 0; d < 64; ++d) s += qr[d] * rot[((size_t)d * 4 + hsh) * 256 + i];
      if ( s > best) { best =  s; bi = i; }
      if (-s > best) { best = -s; bi = i + 256; }
    }
    buckets[((size_t)bh * 4 + hsh) * 2048 + t] = bi;
  }
}

// ---------------------------------------------------------------------------
// Stable counting sort per (bh, hash): keys = bucket*2048 + t are unique and
// hash segments' bucket ranges are disjoint -> global argsort == per-segment
// counting sort, positions offset by hash*2048.
// ---------------------------------------------------------------------------
__global__ __launch_bounds__(256) void sort_kernel(
    const int* __restrict__ buckets, int* __restrict__ sorted_t,
    int* __restrict__ undo)
{
  int bh = blockIdx.x >> 2, hsh = blockIdx.x & 3;
  __shared__ int lb[2048];
  __shared__ int hist[512];
  __shared__ int start[512];
  int tid = threadIdx.x;
  for (int i = tid; i < 2048; i += 256)
    lb[i] = buckets[((size_t)bh * 4 + hsh) * 2048 + i];
  for (int i = tid; i < 512; i += 256) hist[i] = 0;
  __syncthreads();
  for (int i = tid; i < 2048; i += 256) atomicAdd(&hist[lb[i]], 1);
  __syncthreads();
  if (tid == 0) {
    int acc = 0;
    for (int i = 0; i < 512; ++i) { start[i] = acc; acc += hist[i]; }
  }
  __syncthreads();
  for (int b = tid; b < 512; b += 256) {
    int pos = start[b];
    for (int t = 0; t < 2048; ++t) {
      if (lb[t] == b) {
        int s = hsh * 2048 + pos;
        sorted_t[(size_t)bh * 8192 + s] = t;
        undo[(size_t)bh * 8192 + hsh * 2048 + t] = s;
        ++pos;
      }
    }
  }
}

// ---------------------------------------------------------------------------
// Chunked attention: one wave per (bh, chunk). 4 queries x 8 keys (cur+prev
// chunk), keys L2-normalized, self-token masked to -5e4, write out + lse.
// ---------------------------------------------------------------------------
__global__ __launch_bounds__(256) void attn_kernel(
    const float* __restrict__ qk, const float* __restrict__ v,
    const int* __restrict__ sorted_t, float* __restrict__ so,
    float* __restrict__ slog)
{
  int gw   = (blockIdx.x * 256 + threadIdx.x) >> 5;  // 64*2048 waves
  int lane = threadIdx.x & 31;
  int bh = gw >> 11, c = gw & 2047;
  int n = bh >> 2, head = bh & 3;
  int d0 = lane * 2;

  float q0[4], q1[4]; int tq[4];
  #pragma unroll
  for (int i = 0; i < 4; ++i) {
    int s = c * 4 + i;
    int t = sorted_t[(size_t)bh * 8192 + s]; tq[i] = t;
    const float* p = qk + ((size_t)n * 2048 + t) * 256 + head * 64 + d0;
    q0[i] = p[0]; q1[i] = p[1];
  }
  float k0[8], k1[8], v0[8], v1[8]; int tk[8];
  #pragma unroll
  for (int j = 0; j < 8; ++j) {
    int cc = (j < 4) ? c : ((c + 2047) & 2047);     // look-back one chunk
    int s = cc * 4 + (j & 3);
    int t = sorted_t[(size_t)bh * 8192 + s]; tk[j] = t;
    size_t base = ((size_t)n * 2048 + t) * 256 + head * 64 + d0;
    k0[j] = qk[base]; k1[j] = qk[base + 1];
    v0[j] = v[base];  v1[j] = v[base + 1];
  }
  #pragma unroll
  for (int j = 0; j < 8; ++j) {
    float nr = allsum32(k0[j] * k0[j] + k1[j] * k1[j]);
    float sc = 1.f / fmaxf(sqrtf(nr), 1e-12f);
    k0[j] *= sc; k1[j] *= sc;
  }
  float dots[4][8];
  #pragma unroll
  for (int i = 0; i < 4; ++i) {
    #pragma unroll
    for (int j = 0; j < 8; ++j) {
      float p = allsum32(q0[i] * k0[j] + q1[i] * k1[j]) * 0.125f;
      if (tq[i] == tk[j]) p = -5e4f;
      dots[i][j] = p;
    }
  }
  #pragma unroll
  for (int i = 0; i < 4; ++i) {
    float mx = dots[i][0];
    #pragma unroll
    for (int j = 1; j < 8; ++j) mx = fmaxf(mx, dots[i][j]);
    float sum = 0.f;
    #pragma unroll
    for (int j = 0; j < 8; ++j) sum += expf(dots[i][j] - mx);
    float lse = mx + logf(sum);
    float o0 = 0.f, o1 = 0.f;
    #pragma unroll
    for (int j = 0; j < 8; ++j) {
      float w = expf(dots[i][j] - lse);
      o0 += w * v0[j]; o1 += w * v1[j];
    }
    int s = c * 4 + i;
    float* op = so + ((size_t)bh * 8192 + s) * 64 + d0;
    op[0] = o0; op[1] = o1;
    if (lane == 0) slog[(size_t)bh * 8192 + s] = lse;
  }
}

// ---------------------------------------------------------------------------
// Combine hash rounds with softmax over per-round lse; write merged heads.
// attn layout: [N,T,256] with channel = head*64 + d
// ---------------------------------------------------------------------------
__global__ __launch_bounds__(256) void combine_kernel(
    const float* __restrict__ so, const float* __restrict__ slog,
    const int* __restrict__ undo, float* __restrict__ attn)
{
  int gw   = (blockIdx.x * 256 + threadIdx.x) >> 5;
  int lane = threadIdx.x & 31;
  int bh = gw >> 11, t = gw & 2047;
  int n = bh >> 2, head = bh & 3;
  int d0 = lane * 2;
  int s[4]; float lg[4];
  #pragma unroll
  for (int h = 0; h < 4; ++h) {
    s[h]  = undo[(size_t)bh * 8192 + h * 2048 + t];
    lg[h] = slog[(size_t)bh * 8192 + s[h]];
  }
  float mx = fmaxf(fmaxf(lg[0], lg[1]), fmaxf(lg[2], lg[3]));
  float w[4], Z = 0.f;
  #pragma unroll
  for (int h = 0; h < 4; ++h) { w[h] = expf(lg[h] - mx); Z += w[h]; }
  float o0 = 0.f, o1 = 0.f;
  #pragma unroll
  for (int h = 0; h < 4; ++h) {
    const float* p = so + ((size_t)bh * 8192 + s[h]) * 64 + d0;
    float ww = w[h] / Z;
    o0 += ww * p[0]; o1 += ww * p[1];
  }
  float* op = attn + ((size_t)n * 2048 + t) * 256 + head * 64 + d0;
  op[0] = o0; op[1] = o1;
}

// ---------------------------------------------------------------------------
// LayerNorm over last dim (256): one wave per row.
// ---------------------------------------------------------------------------
__global__ __launch_bounds__(256) void ln_kernel(
    const float* __restrict__ in, const float* __restrict__ g,
    const float* __restrict__ b, float* __restrict__ out)
{
  int gw   = (blockIdx.x * 256 + threadIdx.x) >> 5;
  int lane = threadIdx.x & 31;
  const float* x = in + (size_t)gw * 256;
  float v[8];
  #pragma unroll
  for (int i = 0; i < 8; ++i) v[i] = x[lane * 8 + i];
  float s = 0.f;
  #pragma unroll
  for (int i = 0; i < 8; ++i) s += v[i];
  float mean = allsum32(s) * (1.f / 256.f);
  float q = 0.f;
  #pragma unroll
  for (int i = 0; i < 8; ++i) { float d = v[i] - mean; q += d * d; }
  float rs = rsqrtf(allsum32(q) * (1.f / 256.f) + 1e-5f);
  float* o = out + (size_t)gw * 256;
  #pragma unroll
  for (int i = 0; i < 8; ++i) {
    int c = lane * 8 + i;
    o[c] = (v[i] - mean) * rs * g[c] + b[c];
  }
}

// ---------------------------------------------------------------------------
// Final projection to NY=1; output [T, N, 1]
// ---------------------------------------------------------------------------
__global__ __launch_bounds__(256) void proj_kernel(
    const float* __restrict__ hin, const float* __restrict__ pw,
    const float* __restrict__ pb, float* __restrict__ out)
{
  int gw   = (blockIdx.x * 256 + threadIdx.x) >> 5;  // row = n*2048 + t
  int lane = threadIdx.x & 31;
  const float* x = hin + (size_t)gw * 256;
  float s = 0.f;
  #pragma unroll
  for (int i = 0; i < 8; ++i) { int c = lane * 8 + i; s += x[c] * pw[c]; }
  s = allsum32(s);
  if (lane == 0) {
    int t = gw & 2047, n = gw >> 11;
    out[t * 16 + n] = s + pb[0];
  }
}

// ---------------------------------------------------------------------------
extern "C" void kernel_launch(void* const* d_in, const int* in_sizes, int n_in,
                              void* d_out, int out_size, void* d_ws, size_t ws_size,
                              hipStream_t stream) {
  const float* x      = (const float*)d_in[0];
  const float* conv_w = (const float*)d_in[1];
  // d_in[2] = w_time (unused: time marks are zero)
  const float* Wqk    = (const float*)d_in[3];
  const float* Wv     = (const float*)d_in[4];
  const float* Wo     = (const float*)d_in[5];
  const float* bo     = (const float*)d_in[6];
  const float* ln1_g  = (const float*)d_in[7];
  const float* ln1_b  = (const float*)d_in[8];
  const float* W1     = (const float*)d_in[9];
  const float* b1     = (const float*)d_in[10];
  const float* W2     = (const float*)d_in[11];
  const float* b2     = (const float*)d_in[12];
  const float* ln2_g  = (const float*)d_in[13];
  const float* ln2_b  = (const float*)d_in[14];
  const float* rot    = (const float*)d_in[15];
  const float* lnf_g  = (const float*)d_in[16];
  const float* lnf_b  = (const float*)d_in[17];
  const float* proj_w = (const float*)d_in[18];
  const float* proj_b = (const float*)d_in[19];
  float* out = (float*)d_out;

  const size_t MD = 32768ull * 256;        // 8.4M floats
  float* ws   = (float*)d_ws;
  float* h    = ws;                        // [32768,256]
  float* tmp  = h   + MD;                  // [32768,256]
  float* qk   = tmp + MD;                  // [32768,256] (reused as attn-merged)
  float* vv   = qk  + MD;                  // [32768,256]
  float* big  = vv  + MD;                  // 33.6M floats: FFN y1  OR  so[64,8192,64]
  float* slog = big + 32768ull * 1024;     // [64,8192]
  int* buckets  = (int*)(slog + 64ull * 8192);   // [64,4,2048]
  int* sorted_t = buckets + 64 * 4 * 2048;       // [64,8192]
  int* undo     = sorted_t + 64 * 8192;          // [64,8192]

  embed_kernel<<<32768, 256, 0, stream>>>(x, conv_w, h);

  dim3 gD(256, 4);     // N=256 GEMMs, 128-row tiles
  dim3 gF1(256, 16);   // N=1024 GEMM
  for (int l = 0; l < 2; ++l) {
    const float* Wqk_l = Wqk + (size_t)l * 256 * 256;
    const float* Wv_l  = Wv  + (size_t)l * 256 * 256;
    const float* Wo_l  = Wo  + (size_t)l * 256 * 256;
    const float* bo_l  = bo  + (size_t)l * 256;
    const float* W1_l  = W1  + (size_t)l * 256 * 1024;
    const float* b1_l  = b1  + (size_t)l * 1024;
    const float* W2_l  = W2  + (size_t)l * 1024 * 256;
    const float* b2_l  = b2  + (size_t)l * 256;
    const float* rot_l = rot + (size_t)l * 64 * 4 * 256;

    gemm_wmma<false, false, false><<<gD, 256, 0, stream>>>(h, Wqk_l, nullptr, nullptr, qk, 32768, 256, 256);
    gemm_wmma<false, false, false><<<gD, 256, 0, stream>>>(h, Wv_l,  nullptr, nullptr, vv, 32768, 256, 256);

    hash_kernel<<<512, 256, 0, stream>>>(qk, rot_l, buckets);
    sort_kernel<<<256, 256, 0, stream>>>(buckets, sorted_t, undo);
    attn_kernel<<<16384, 256, 0, stream>>>(qk, vv, sorted_t, big, slog);
    combine_kernel<<<16384, 256, 0, stream>>>(big, slog, undo, qk);  // qk := merged attn

    gemm_wmma<true, false, true><<<gD, 256, 0, stream>>>(qk, Wo_l, bo_l, h, tmp, 32768, 256, 256);
    ln_kernel<<<4096, 256, 0, stream>>>(tmp, ln1_g + l * 256, ln1_b + l * 256, h);

    gemm_wmma<true, true, false><<<gF1, 256, 0, stream>>>(h, W1_l, b1_l, nullptr, big, 32768, 1024, 256);
    gemm_wmma<true, false, true><<<gD, 256, 0, stream>>>(big, W2_l, b2_l, h, tmp, 32768, 256, 1024);
    ln_kernel<<<4096, 256, 0, stream>>>(tmp, ln2_g + l * 256, ln2_b + l * 256, h);
  }

  ln_kernel<<<4096, 256, 0, stream>>>(h, lnf_g, lnf_b, tmp);
  proj_kernel<<<4096, 256, 0, stream>>>(tmp, proj_w, proj_b, out);
}
